// _AGSITMLP_2044404433080
// MI455X (gfx1250) — matrix operand order repaired and checked by reading, not compiled, not run on hardware
//
#include <hip/hip_runtime.h>

// ---------------------------------------------------------------------------
// MI455X (gfx1250) implementation. wave32, WMMA f16->f32 for all GEMMs.
// ~71 GFLOP vs ~44MB traffic -> compute bound. Each wave computes a 16x64
// output strip: 8x ds_load_b128 (B frags) issued together, then 4 back-to-back
// v_wmma under a single s_wait_dscnt.
// ---------------------------------------------------------------------------

typedef _Float16 half_t;
typedef __attribute__((ext_vector_type(16))) _Float16 v16h;
typedef __attribute__((ext_vector_type(8)))  float    v8f;

#define WMMA_F16(a, b, c) \
  __builtin_amdgcn_wmma_f32_16x16x32_f16(false, (a), false, (b), (short)0, (c), false, false)

#define B_SZ     8192
#define DNUM     1024
#define DTOK     512
#define TDIM     64
#define NHEADS   4
#define TOPK     64
#define HDIM     16
#define FFND     128
#define DELTA_IN 576
#define DELTA_HID 144
#define BETA_IN  516
#define BETA_HID 128

// ---------------------------------------------------------------------------
// Generic WMMA GEMM: out = epilogue(A[MxK] @ B[KxN] + bias) (+ residual)
// Block tile 128x64 (8 waves stacked in M, each wave owns 16 rows x 64 cols,
// i.e. 4 accumulators). K-step 32. Vectorized staging fast path; scalar
// bounds-checked slow path for edge tiles (K=516, N=144 tail).
// act: 0=none 1=relu 2=sigmoid 3=gelu(exact, erf)
// ---------------------------------------------------------------------------
#define BM 128
#define BN 64
#define BKK 32
#define LPAD 8   // row pitch 40 halves = 80B (16B aligned, conflict-free)

__global__ __launch_bounds__(256) void gemm_wmma_kernel(
    const half_t* __restrict__ A, const half_t* __restrict__ Bw,
    const float* __restrict__ bias, const float* __restrict__ resid,
    float* __restrict__ Cf, half_t* __restrict__ Ch,
    int M, int N, int K, int act)
{
  __shared__ alignas(16) half_t As[BM][BKK + LPAD];
  __shared__ alignas(16) half_t Bs[BN][BKK + LPAD];   // transposed: Bs[n][k]

  const int tid  = threadIdx.x;
  const int wid  = tid >> 5;        // 0..7 : 16-row strip within the block
  const int lane = tid & 31;
  const int hf   = lane >> 4;       // half-wave
  const int l16  = lane & 15;
  const int bm0  = blockIdx.x * BM;
  const int bn0  = blockIdx.y * BN;

  const bool fastA = ((K & 7) == 0) && (bm0 + BM <= M);
  const bool fastBn = (bn0 + BN <= N) && ((N & 3) == 0);

  v8f acc0 = {}, acc1 = {}, acc2 = {}, acc3 = {};

  for (int k0 = 0; k0 < K; k0 += BKK) {
    const bool kin = (k0 + BKK <= K);
    // ---- stage A tile (128x32 f16) ----
    if (fastA && kin) {
      // 512 16B chunks, 2 per thread: fully vectorized b128 path
      #pragma unroll
      for (int c = tid; c < (BM * BKK) / 8; c += 256) {
        int row = c >> 2, c8 = (c & 3) * 8;
        *(uint4*)&As[row][c8] =
            *(const uint4*)(A + (size_t)(bm0 + row) * K + k0 + c8);
      }
    } else {
      for (int i = tid; i < BM * BKK; i += 256) {
        int r = i >> 5, c = i & (BKK - 1);
        int gm = bm0 + r, gk = k0 + c;
        As[r][c] = (gm < M && gk < K) ? A[(size_t)gm * K + gk] : (half_t)0;
      }
    }
    // ---- stage B tile transposed (32x64 -> Bs[n][k]) ----
    if (fastBn && kin) {
      // 256 chunks of (2 K-rows x 4 N-cols): b64 loads, b32 transposed stores
      int kp = tid >> 4;            // 0..15 -> k = 2*kp
      int n4 = (tid & 15) * 4;      // n base
      int k  = kp * 2;
      union { uint2 v; unsigned short u[4]; } ra, rb;
      ra.v = *(const uint2*)(Bw + (size_t)(k0 + k) * N + bn0 + n4);
      rb.v = *(const uint2*)(Bw + (size_t)(k0 + k + 1) * N + bn0 + n4);
      #pragma unroll
      for (int j = 0; j < 4; ++j)
        *(unsigned int*)&Bs[n4 + j][k] =
            (unsigned int)ra.u[j] | ((unsigned int)rb.u[j] << 16);
    } else {
      for (int i = tid; i < BKK * BN; i += 256) {
        int r = i >> 6, c = i & (BN - 1);
        int gk = k0 + r, gn = bn0 + c;
        Bs[c][r] = (gk < K && gn < N) ? Bw[(size_t)gk * N + gn] : (half_t)0;
      }
    }
    if (k0 + 2 * BKK <= K) {   // prefetch next K tile -> global_prefetch_b8
      __builtin_prefetch(&A[(size_t)(bm0 + (tid >> 1)) * K + k0 + BKK], 0, 1);
      __builtin_prefetch(&Bw[(size_t)(k0 + BKK + wid) * N + bn0], 0, 1);
    }
    __syncthreads();

    // A fragment (shared across the 4 column tiles): 2x ds_load_b128
    v16h af;
    const int am = wid * 16 + l16;
    #pragma unroll
    for (int r = 0; r < 8; ++r) {
      int kk = (r >> 2) * 16 + hf * 8 + (r & 3) * 2;
      af[2 * r]     = As[am][kk];
      af[2 * r + 1] = As[am][kk + 1];
    }
    // All 4 B fragments loaded up front (8x ds_load_b128, one wait), then
    // 4 back-to-back WMMAs so the matrix pipe stays busy.
    v16h bf0, bf1, bf2, bf3;
    #pragma unroll
    for (int r = 0; r < 8; ++r) {
      int kk = (r >> 2) * 16 + hf * 8 + (r & 3) * 2;
      bf0[2 * r] = Bs[ 0 + l16][kk]; bf0[2 * r + 1] = Bs[ 0 + l16][kk + 1];
      bf1[2 * r] = Bs[16 + l16][kk]; bf1[2 * r + 1] = Bs[16 + l16][kk + 1];
      bf2[2 * r] = Bs[32 + l16][kk]; bf2[2 * r + 1] = Bs[32 + l16][kk + 1];
      bf3[2 * r] = Bs[48 + l16][kk]; bf3[2 * r + 1] = Bs[48 + l16][kk + 1];
    }
    acc0 = WMMA_F16(af, bf0, acc0);
    acc1 = WMMA_F16(af, bf1, acc1);
    acc2 = WMMA_F16(af, bf2, acc2);
    acc3 = WMMA_F16(af, bf3, acc3);
    __syncthreads();
  }

  // epilogue: C/D layout -> lane row = hf*8 + r, col = lane&15
  #pragma unroll
  for (int nt = 0; nt < 4; ++nt) {
    const v8f& acc = (nt == 0) ? acc0 : (nt == 1) ? acc1 : (nt == 2) ? acc2 : acc3;
    #pragma unroll
    for (int r = 0; r < 8; ++r) {
      int gm = bm0 + wid * 16 + hf * 8 + r;
      int gn = bn0 + nt * 16 + l16;
      if (gm < M && gn < N) {
        float v = acc[r];
        if (bias) v += bias[gn];
        if (act == 1)      v = fmaxf(v, 0.f);
        else if (act == 2) v = 1.f / (1.f + __expf(-v));
        else if (act == 3) v = 0.5f * v * (1.f + erff(v * 0.70710678118654752f));
        if (resid) v += resid[(size_t)gm * N + gn];
        if (Cf) Cf[(size_t)gm * N + gn] = v;
        if (Ch) Ch[(size_t)gm * N + gn] = (half_t)v;
      }
    }
  }
}

// ---------------------------------------------------------------------------
// Attention: one block per (b, head). 128 threads = 4 waves.
// scores = (Q K^T)/4 + gate_bias[j]; softmax over j; ao = attn @ V.
// HEAD_DIM=16 is zero-padded to K=32 for the score WMMA.
// ---------------------------------------------------------------------------
__global__ __launch_bounds__(128) void attention_kernel(
    const half_t* __restrict__ qg, const half_t* __restrict__ kg,
    const half_t* __restrict__ vg, const float* __restrict__ tkg,
    half_t* __restrict__ aog)
{
  __shared__ half_t qs[TOPK][HDIM];
  __shared__ half_t ks[TOPK][HDIM];
  __shared__ half_t vs[TOPK][HDIM];
  __shared__ float  sc[TOPK][TOPK + 1];
  __shared__ half_t at[TOPK][TOPK];
  __shared__ float  gbias[TOPK];

  const int b = blockIdx.x >> 2;
  const int h = blockIdx.x & 3;
  const int tid = threadIdx.x;
  const size_t base = ((size_t)b * TOPK) * TDIM + h * HDIM;

  for (int i = tid; i < TOPK * HDIM; i += 128) {
    int tok = i >> 4, d = i & 15;
    size_t a = base + (size_t)tok * TDIM + d;
    qs[tok][d] = qg[a];
    ks[tok][d] = kg[a];
    vs[tok][d] = vg[a];
  }
  if (tid < TOPK) gbias[tid] = tkg[(size_t)b * TOPK + tid];
  __syncthreads();

  const int wid = tid >> 5, lane = tid & 31, hf = lane >> 4, l16 = lane & 15;

  // 16 score tiles over (it, jt) in 4x4; each wave does 4.
  for (int t = wid; t < 16; t += 4) {
    int it = t >> 2, jt = t & 3;
    v16h af = {}, bf = {};
    #pragma unroll
    for (int r = 0; r < 8; ++r) {
      int kk = (r >> 2) * 16 + hf * 8 + (r & 3) * 2;
      if (kk < HDIM) {                       // K 16..31 stays zero
        af[2 * r]     = qs[it * 16 + l16][kk];
        af[2 * r + 1] = qs[it * 16 + l16][kk + 1];
      }
    }
    #pragma unroll
    for (int r = 0; r < 8; ++r) {
      int kk = (r >> 2) * 16 + hf * 8 + (r & 3) * 2;
      if (kk < HDIM) {
        bf[2 * r]     = ks[jt * 16 + l16][kk];
        bf[2 * r + 1] = ks[jt * 16 + l16][kk + 1];
      }
    }
    v8f c = {};
    c = WMMA_F16(af, bf, c);
    #pragma unroll
    for (int r = 0; r < 8; ++r) {
      int row = it * 16 + hf * 8 + r;
      int col = jt * 16 + l16;
      sc[row][col] = c[r] * 0.25f + gbias[col];   // 1/sqrt(HEAD_DIM)=0.25
    }
  }
  __syncthreads();

  if (tid < TOPK) {   // softmax, one row per thread
    float mx = -1e30f;
    for (int j = 0; j < TOPK; ++j) mx = fmaxf(mx, sc[tid][j]);
    float s = 0.f;
    for (int j = 0; j < TOPK; ++j) { float e = __expf(sc[tid][j] - mx); sc[tid][j] = e; s += e; }
    float inv = 1.f / s;
    for (int j = 0; j < TOPK; ++j) at[tid][j] = (half_t)(sc[tid][j] * inv);
  }
  __syncthreads();

  // ao = attn(64x64) @ V(64x16): 4 row tiles, K=64 in two chunks.
  // Both chunks' fragments loaded up front, then chained WMMAs.
  {
    int it = wid;
    v16h af0, bf0c, af1, bf1c;
    #pragma unroll
    for (int r = 0; r < 8; ++r) {
      int kk = (r >> 2) * 16 + hf * 8 + (r & 3) * 2;
      af0[2 * r]     = at[it * 16 + l16][kk];
      af0[2 * r + 1] = at[it * 16 + l16][kk + 1];
      af1[2 * r]     = at[it * 16 + l16][32 + kk];
      af1[2 * r + 1] = at[it * 16 + l16][32 + kk + 1];
    }
    #pragma unroll
    for (int r = 0; r < 8; ++r) {
      int kk = (r >> 2) * 16 + hf * 8 + (r & 3) * 2;
      bf0c[2 * r]     = vs[kk][l16];
      bf0c[2 * r + 1] = vs[kk + 1][l16];
      bf1c[2 * r]     = vs[32 + kk][l16];
      bf1c[2 * r + 1] = vs[32 + kk + 1][l16];
    }
    v8f c = {};
    c = WMMA_F16(af0, bf0c, c);
    c = WMMA_F16(af1, bf1c, c);
    #pragma unroll
    for (int r = 0; r < 8; ++r) {
      int tok = it * 16 + hf * 8 + r;
      aog[base + (size_t)tok * TDIM + l16] = (half_t)c[r];
    }
  }
}

// ---------------------------------------------------------------------------
// LayerNorm over rows of length L, one wave32 per row. Writes f16.
// ---------------------------------------------------------------------------
__global__ void ln_rows_kernel(const float* __restrict__ src,
                               const float* __restrict__ g,
                               const float* __restrict__ bb,
                               half_t* __restrict__ dst, int rows, int L)
{
  int row = blockIdx.x * (blockDim.x >> 5) + (threadIdx.x >> 5);
  if (row >= rows) return;
  int lane = threadIdx.x & 31;
  const float* x = src + (size_t)row * L;
  float s = 0.f, s2 = 0.f;
  for (int i = lane; i < L; i += 32) { float v = x[i]; s += v; s2 += v * v; }
  for (int o = 16; o; o >>= 1) { s += __shfl_xor(s, o, 32); s2 += __shfl_xor(s2, o, 32); }
  float mean = s / L;
  float var  = s2 / L - mean * mean;
  float inv  = rsqrtf(var + 1e-5f);
  half_t* d = dst + (size_t)row * L;
  for (int i = lane; i < L; i += 32)
    d[i] = (half_t)((x[i] - mean) * inv * g[i] + bb[i]);
}

// ---------------------------------------------------------------------------
// Gate stats (mean/max/pop-std) + iterative top-64 (desc, lower index on tie).
// One block of 256 threads per batch row.
// ---------------------------------------------------------------------------
__global__ __launch_bounds__(256) void stats_topk_kernel(
    const float* __restrict__ gate, float* __restrict__ stats,
    float* __restrict__ tkg, float* __restrict__ tkw, int* __restrict__ tki)
{
  __shared__ float vals[DNUM];
  __shared__ float red[256];
  __shared__ int   redi[256];
  __shared__ float topv[TOPK];

  const int b = blockIdx.x, tid = threadIdx.x;
  const float* g = gate + (size_t)b * DNUM;
  float s = 0.f, s2 = 0.f, mx = -1e30f;
  for (int i = tid; i < DNUM; i += 256) {
    float v = g[i]; vals[i] = v; s += v; s2 += v * v; mx = fmaxf(mx, v);
  }
  red[tid] = s;  __syncthreads();
  for (int o = 128; o; o >>= 1) { if (tid < o) red[tid] += red[tid + o]; __syncthreads(); }
  float sum = red[0]; __syncthreads();
  red[tid] = s2; __syncthreads();
  for (int o = 128; o; o >>= 1) { if (tid < o) red[tid] += red[tid + o]; __syncthreads(); }
  float sum2 = red[0]; __syncthreads();
  red[tid] = mx; __syncthreads();
  for (int o = 128; o; o >>= 1) { if (tid < o) red[tid] = fmaxf(red[tid], red[tid + o]); __syncthreads(); }
  float gmax = red[0]; __syncthreads();
  if (tid == 0) {
    float mean = sum / DNUM;
    float var  = sum2 / DNUM - mean * mean;
    stats[b * 3 + 0] = mean;
    stats[b * 3 + 1] = gmax;
    stats[b * 3 + 2] = sqrtf(fmaxf(var, 0.f));
  }

  float topsum = 0.f;
  for (int it = 0; it < TOPK; ++it) {
    float bv = -1e30f; int bi = DNUM;
    for (int i = tid; i < DNUM; i += 256) {
      float v = vals[i];
      if (v > bv || (v == bv && i < bi)) { bv = v; bi = i; }
    }
    red[tid] = bv; redi[tid] = bi; __syncthreads();
    for (int o = 128; o; o >>= 1) {
      if (tid < o) {
        float ov = red[tid + o]; int oi = redi[tid + o];
        if (ov > red[tid] || (ov == red[tid] && oi < redi[tid])) { red[tid] = ov; redi[tid] = oi; }
      }
      __syncthreads();
    }
    float v0 = red[0]; int i0 = redi[0];
    __syncthreads();
    if (tid == 0) {
      tkg[(size_t)b * TOPK + it] = v0;
      tki[(size_t)b * TOPK + it] = i0;
      topv[it] = v0;
      vals[i0] = -1e30f;
    }
    topsum += v0;
    __syncthreads();
  }
  if (tid < TOPK)
    tkw[(size_t)b * TOPK + tid] = topv[tid] / (topsum + 1e-6f);
}

// ---------------------------------------------------------------------------
// Small elementwise / reduction helpers
// ---------------------------------------------------------------------------
__global__ void f32_to_f16_kernel(const float* __restrict__ s, half_t* __restrict__ d, size_t n) {
  size_t i = (size_t)blockIdx.x * 256 + threadIdx.x;
  if (i < n) d[i] = (half_t)s[i];
}

__global__ void gate_mul_kernel(const float* __restrict__ x, const float* __restrict__ g,
                                half_t* __restrict__ d, size_t n) {
  size_t i = (size_t)blockIdx.x * 256 + threadIdx.x;
  if (i < n) d[i] = (half_t)(x[i] * g[i]);
}

__global__ void tokenize_kernel(const float* __restrict__ x, const float* __restrict__ vp,
                                const float* __restrict__ fe, const float* __restrict__ tkw,
                                const int* __restrict__ tki, float* __restrict__ tokens) {
  size_t i = (size_t)blockIdx.x * 256 + threadIdx.x;
  if (i >= (size_t)B_SZ * TOPK * TDIM) return;
  int d = (int)(i % TDIM);
  size_t tk = i / TDIM;              // b*64 + k
  int b = (int)(tk / TOPK);
  int idx = tki[tk];
  float w  = tkw[tk];
  float xv = x[(size_t)b * DNUM + idx];
  tokens[i] = (xv * vp[(size_t)idx * TDIM + d] + fe[(size_t)idx * TDIM + d]) * w;
}

__global__ void zint_kernel(const float* __restrict__ x2, float* __restrict__ z) {
  int b = blockIdx.x, t = threadIdx.x;  // 64 threads
  float s = 0.f;
  for (int i = 0; i < TOPK; ++i) s += x2[((size_t)b * TOPK + i) * TDIM + t];
  z[(size_t)b * TDIM + t] = s * (1.0f / TOPK);
}

__global__ void concat_din_kernel(const float* __restrict__ hb, const float* __restrict__ z,
                                  float* __restrict__ dst) {
  size_t i = (size_t)blockIdx.x * 256 + threadIdx.x;
  if (i >= (size_t)B_SZ * DELTA_IN) return;
  int b = (int)(i / DELTA_IN), c = (int)(i % DELTA_IN);
  dst[i] = (c < DTOK) ? hb[(size_t)b * DTOK + c] : z[(size_t)b * TDIM + (c - DTOK)];
}

__global__ void concat_bin_kernel(const float* __restrict__ hb, const float* __restrict__ yb,
                                  const float* __restrict__ st, float* __restrict__ dst) {
  size_t i = (size_t)blockIdx.x * 256 + threadIdx.x;
  if (i >= (size_t)B_SZ * BETA_IN) return;
  int b = (int)(i / BETA_IN), c = (int)(i % BETA_IN);
  float v;
  if (c < DTOK)       v = hb[(size_t)b * DTOK + c];
  else if (c == DTOK) v = yb[b];
  else                v = st[b * 3 + (c - DTOK - 1)];   // mean, max, std
  dst[i] = v;
}

__global__ void ybase_kernel(const float* __restrict__ hb, const float* __restrict__ Wb,
                             const float* __restrict__ bb, float* __restrict__ yb) {
  int row = blockIdx.x * (blockDim.x >> 5) + (threadIdx.x >> 5);
  if (row >= B_SZ) return;
  int lane = threadIdx.x & 31;
  float s = 0.f;
  for (int i = lane; i < DTOK; i += 32) s += hb[(size_t)row * DTOK + i] * Wb[i];
  for (int o = 16; o; o >>= 1) s += __shfl_xor(s, o, 32);
  if (lane == 0) yb[row] = s + bb[0];
}

__global__ void final_kernel(const float* __restrict__ yb,
                             const float* __restrict__ dh, const float* __restrict__ Wd2,
                             const float* __restrict__ bd2,
                             const float* __restrict__ bh, const float* __restrict__ Wbe2,
                             const float* __restrict__ bbe2, float* __restrict__ out) {
  int row = blockIdx.x * (blockDim.x >> 5) + (threadIdx.x >> 5);
  if (row >= B_SZ) return;
  int lane = threadIdx.x & 31;
  float sd = 0.f, sb = 0.f;
  for (int i = lane; i < DELTA_HID; i += 32) sd += dh[(size_t)row * DELTA_HID + i] * Wd2[i];
  for (int i = lane; i < BETA_HID;  i += 32) sb += bh[(size_t)row * BETA_HID  + i] * Wbe2[i];
  for (int o = 16; o; o >>= 1) { sd += __shfl_xor(sd, o, 32); sb += __shfl_xor(sb, o, 32); }
  if (lane == 0) {
    float delta = sd + bd2[0];
    float beta  = 1.f / (1.f + __expf(-(sb + bbe2[0])));
    out[row] = yb[row] + beta * delta;
  }
}

// ---------------------------------------------------------------------------
// Host orchestration
// ---------------------------------------------------------------------------
enum {
  IN_X = 0, IN_WG, IN_BG, IN_WB1, IN_BB1, IN_VPROJ, IN_FEMB,
  IN_LN1G, IN_LN1B, IN_WQ, IN_BQ, IN_WK, IN_BKb, IN_WV, IN_BV,
  IN_WO, IN_BO, IN_LN2G, IN_LN2B, IN_WF1, IN_BF1, IN_WF2, IN_BF2,
  IN_WBASE, IN_BBASE, IN_LNDG, IN_LNDB, IN_WD1, IN_BD1, IN_WD2, IN_BD2,
  IN_LNBG, IN_LNBB, IN_WBE1, IN_BBE1, IN_WBE2, IN_BBE2
};

extern "C" void kernel_launch(void* const* d_in, const int* in_sizes, int n_in,
                              void* d_out, int out_size, void* d_ws, size_t ws_size,
                              hipStream_t stream)
{
  (void)in_sizes; (void)n_in; (void)out_size; (void)ws_size;
  const float* x = (const float*)d_in[IN_X];

  char* ws = (char*)d_ws;
  size_t off = 0;
  auto alloc = [&](size_t bytes) -> void* {
    void* p = ws + off;
    off += (bytes + 255) & ~(size_t)255;
    return p;
  };

  const size_t NT = (size_t)B_SZ * TOPK;     // total tokens: 524288

  // f16 shadows
  half_t* x_h    = (half_t*)alloc((size_t)B_SZ * DNUM * 2);
  half_t* Wg_h   = (half_t*)alloc((size_t)DNUM * DNUM * 2);
  half_t* Wb1_h  = (half_t*)alloc((size_t)DNUM * DTOK * 2);
  half_t* Wq_h   = (half_t*)alloc((size_t)TDIM * TDIM * 2);
  half_t* Wk_h   = (half_t*)alloc((size_t)TDIM * TDIM * 2);
  half_t* Wv_h   = (half_t*)alloc((size_t)TDIM * TDIM * 2);
  half_t* Wo_h   = (half_t*)alloc((size_t)TDIM * TDIM * 2);
  half_t* Wf1_h  = (half_t*)alloc((size_t)TDIM * FFND * 2);
  half_t* Wf2_h  = (half_t*)alloc((size_t)FFND * TDIM * 2);
  half_t* Wd1_h  = (half_t*)alloc((size_t)DELTA_IN * DELTA_HID * 2);
  half_t* Wbe1_h = (half_t*)alloc((size_t)BETA_IN * BETA_HID * 2);
  // f32 / f16 intermediates
  float*  gate   = (float*)alloc((size_t)B_SZ * DNUM * 4);
  half_t* xg_h   = (half_t*)alloc((size_t)B_SZ * DNUM * 2);
  float*  hbase  = (float*)alloc((size_t)B_SZ * DTOK * 4);
  float*  stats  = (float*)alloc((size_t)B_SZ * 3 * 4);
  float*  tkg    = (float*)alloc(NT * 4);
  float*  tkw    = (float*)alloc(NT * 4);
  int*    tki    = (int*)alloc(NT * 4);
  float*  ybase  = (float*)alloc((size_t)B_SZ * 4);
  float*  tokens = (float*)alloc(NT * TDIM * 4);
  half_t* xln_h  = (half_t*)alloc(NT * TDIM * 2);
  half_t* q_h    = (half_t*)alloc(NT * TDIM * 2);
  half_t* k_h    = (half_t*)alloc(NT * TDIM * 2);
  half_t* v_h    = (half_t*)alloc(NT * TDIM * 2);
  half_t* ao_h   = (half_t*)alloc(NT * TDIM * 2);
  float*  x1     = (float*)alloc(NT * TDIM * 4);
  half_t* h2_h   = (half_t*)alloc(NT * TDIM * 2);
  half_t* ffn_h  = (half_t*)alloc(NT * FFND * 2);
  float*  x2     = (float*)alloc(NT * TDIM * 4);
  float*  zint   = (float*)alloc((size_t)B_SZ * TDIM * 4);
  float*  dinr   = (float*)alloc((size_t)B_SZ * DELTA_IN * 4);
  half_t* din_h  = (half_t*)alloc((size_t)B_SZ * DELTA_IN * 2);
  float*  dh     = (float*)alloc((size_t)B_SZ * DELTA_HID * 4);
  float*  binr   = (float*)alloc((size_t)B_SZ * BETA_IN * 4);
  half_t* bin_h  = (half_t*)alloc((size_t)B_SZ * BETA_IN * 2);
  float*  bh     = (float*)alloc((size_t)B_SZ * BETA_HID * 4);

  auto cvt = [&](int idx, half_t* dst, size_t n) {
    f32_to_f16_kernel<<<(unsigned)((n + 255) / 256), 256, 0, stream>>>(
        (const float*)d_in[idx], dst, n);
  };
  auto gemm = [&](const half_t* A, const half_t* Bm, int bidx, const float* resid,
                  float* Cf, half_t* Ch, int M, int N, int K, int act) {
    dim3 grid((M + BM - 1) / BM, (N + BN - 1) / BN);
    gemm_wmma_kernel<<<grid, 256, 0, stream>>>(A, Bm, (const float*)d_in[bidx],
                                               resid, Cf, Ch, M, N, K, act);
  };

  // --- convert inputs / weights to f16 ---
  cvt(IN_X,    x_h,    (size_t)B_SZ * DNUM);
  cvt(IN_WG,   Wg_h,   (size_t)DNUM * DNUM);
  cvt(IN_WB1,  Wb1_h,  (size_t)DNUM * DTOK);
  cvt(IN_WQ,   Wq_h,   (size_t)TDIM * TDIM);
  cvt(IN_WK,   Wk_h,   (size_t)TDIM * TDIM);
  cvt(IN_WV,   Wv_h,   (size_t)TDIM * TDIM);
  cvt(IN_WO,   Wo_h,   (size_t)TDIM * TDIM);
  cvt(IN_WF1,  Wf1_h,  (size_t)TDIM * FFND);
  cvt(IN_WF2,  Wf2_h,  (size_t)FFND * TDIM);
  cvt(IN_WD1,  Wd1_h,  (size_t)DELTA_IN * DELTA_HID);
  cvt(IN_WBE1, Wbe1_h, (size_t)BETA_IN * BETA_HID);

  // --- gate = sigmoid(x @ Wg + bg) ---
  gemm(x_h, Wg_h, IN_BG, nullptr, gate, nullptr, B_SZ, DNUM, DNUM, 2);

  // --- gate stats + top-64 ---
  stats_topk_kernel<<<B_SZ, 256, 0, stream>>>(gate, stats, tkg, tkw, tki);

  // --- h_base = relu((x*gate) @ Wb1 + bb1) ---
  gate_mul_kernel<<<(unsigned)(((size_t)B_SZ * DNUM + 255) / 256), 256, 0, stream>>>(
      x, gate, xg_h, (size_t)B_SZ * DNUM);
  gemm(xg_h, Wb1_h, IN_BB1, nullptr, hbase, nullptr, B_SZ, DTOK, DNUM, 1);
  ybase_kernel<<<B_SZ / 8, 256, 0, stream>>>(hbase, (const float*)d_in[IN_WBASE],
                                             (const float*)d_in[IN_BBASE], ybase);

  // --- tokenize + LN1 ---
  tokenize_kernel<<<(unsigned)((NT * TDIM + 255) / 256), 256, 0, stream>>>(
      x, (const float*)d_in[IN_VPROJ], (const float*)d_in[IN_FEMB], tkw, tki, tokens);
  ln_rows_kernel<<<(unsigned)((NT + 7) / 8), 256, 0, stream>>>(
      tokens, (const float*)d_in[IN_LN1G], (const float*)d_in[IN_LN1B], xln_h, (int)NT, TDIM);

  // --- QKV projections ---
  gemm(xln_h, Wq_h, IN_BQ,  nullptr, nullptr, q_h, (int)NT, TDIM, TDIM, 0);
  gemm(xln_h, Wk_h, IN_BKb, nullptr, nullptr, k_h, (int)NT, TDIM, TDIM, 0);
  gemm(xln_h, Wv_h, IN_BV,  nullptr, nullptr, v_h, (int)NT, TDIM, TDIM, 0);

  // --- attention (one block per (b, head)) ---
  attention_kernel<<<B_SZ * NHEADS, 128, 0, stream>>>(q_h, k_h, v_h, tkg, ao_h);

  // --- x1 = tokens + ao @ Wo + bo ---
  gemm(ao_h, Wo_h, IN_BO, tokens, x1, nullptr, (int)NT, TDIM, TDIM, 0);

  // --- FFN: x2 = x1 + gelu(LN2(x1) @ Wf1 + bf1) @ Wf2 + bf2 ---
  ln_rows_kernel<<<(unsigned)((NT + 7) / 8), 256, 0, stream>>>(
      x1, (const float*)d_in[IN_LN2G], (const float*)d_in[IN_LN2B], h2_h, (int)NT, TDIM);
  gemm(h2_h, Wf1_h, IN_BF1, nullptr, nullptr, ffn_h, (int)NT, FFND, TDIM, 3);
  gemm(ffn_h, Wf2_h, IN_BF2, x1, x2, nullptr, (int)NT, TDIM, FFND, 0);

  // --- z_int = mean over tokens ---
  zint_kernel<<<B_SZ, TDIM, 0, stream>>>(x2, zint);

  // --- delta head ---
  concat_din_kernel<<<(unsigned)(((size_t)B_SZ * DELTA_IN + 255) / 256), 256, 0, stream>>>(
      hbase, zint, dinr);
  ln_rows_kernel<<<B_SZ / 8, 256, 0, stream>>>(
      dinr, (const float*)d_in[IN_LNDG], (const float*)d_in[IN_LNDB], din_h, B_SZ, DELTA_IN);
  gemm(din_h, Wd1_h, IN_BD1, nullptr, dh, nullptr, B_SZ, DELTA_HID, DELTA_IN, 1);

  // --- beta head ---
  concat_bin_kernel<<<(unsigned)(((size_t)B_SZ * BETA_IN + 255) / 256), 256, 0, stream>>>(
      hbase, ybase, stats, binr);
  ln_rows_kernel<<<B_SZ / 8, 256, 0, stream>>>(
      binr, (const float*)d_in[IN_LNBG], (const float*)d_in[IN_LNBB], bin_h, B_SZ, BETA_IN);
  gemm(bin_h, Wbe1_h, IN_BBE1, nullptr, bh, nullptr, B_SZ, BETA_HID, BETA_IN, 1);

  // --- out = y_base + beta * delta ---
  final_kernel<<<B_SZ / 8, 256, 0, stream>>>(
      ybase, dh, (const float*)d_in[IN_WD2], (const float*)d_in[IN_BD2],
      bh, (const float*)d_in[IN_WBE2], (const float*)d_in[IN_BBE2], (float*)d_out);
}
